// BiLSTM_CRF_83476984365127
// MI455X (gfx1250) — matrix operand order repaired and checked by reading, not compiled, not run on hardware
//
#include <hip/hip_runtime.h>
#include <hip/hip_bf16.h>
#include <math.h>

typedef __bf16 bf16_t;
typedef __attribute__((ext_vector_type(16))) __bf16 bf16x16;
typedef __attribute__((ext_vector_type(8)))  __bf16 bf16x8;
typedef __attribute__((ext_vector_type(8)))  float  f32x8;

#define SEQ   16384
#define EMB   512
#define HDIM  512            // per-direction hidden
#define G4    2048           // 4*HDIM
#define NCLS  32
#define NIT   5

#if defined(__has_builtin)
#if __has_builtin(__builtin_amdgcn_s_wait_asynccnt)
#define WAIT_ASYNC(n) __builtin_amdgcn_s_wait_asynccnt(n)
#endif
#endif
#ifndef WAIT_ASYNC
#define WAIT_ASYNC(n) asm volatile("s_wait_asynccnt " #n ::: "memory")
#endif

__device__ __forceinline__ f32x8 wmma_bf16(bf16x16 a, bf16x16 b, f32x8 c) {
    return __builtin_amdgcn_wmma_f32_16x16x32_bf16(
        /*neg_a=*/false, a, /*neg_b=*/false, b,
        /*c_mod=*/(short)0, c, /*reuse_a=*/false, /*reuse_b=*/false);
}

// Async global->LDS 16B copy (per active lane). VDST = LDS byte address
// (low 32 bits of the flat shared address), VADDR = 64-bit global address.
__device__ __forceinline__ void async_load_b128(void* lds_dst, const void* gsrc) {
    unsigned lo = (unsigned)(size_t)lds_dst;
    unsigned long long ga = (unsigned long long)(size_t)gsrc;
    asm volatile("global_load_async_to_lds_b128 %0, %1, off"
                 :: "v"(lo), "v"(ga) : "memory");
}

// ---------------------------------------------------------------- converts
__global__ void cvt_f32_bf16_kernel(const float* __restrict__ src,
                                    bf16_t* __restrict__ dst, int n) {
    int i = blockIdx.x * 256 + threadIdx.x;
    if (i < n) dst[i] = (bf16_t)src[i];
}

__global__ void cvt_batch_kernel(const float* __restrict__ x,
                                 bf16_t* __restrict__ xb,
                                 bf16_t* __restrict__ xrev) {
    int i = blockIdx.x * 256 + threadIdx.x;   // i < SEQ*EMB
    int t = i >> 9;                           // /EMB
    int k = i & 511;
    float v = x[i];
    xb[i] = (bf16_t)v;
    xrev[(size_t)(SEQ - 1 - t) * EMB + k] = (bf16_t)v;
}

// ---------------------------------------------------------------- WMMA GEMM
// out[M=SEQ, N=G4] = X[M,K=EMB] * W[N,K]^T + b0[N] + b1[N]
// block = 256 threads (8 waves); block tile M=128, N=64.
// Operands staged global->LDS with async copies (ASYNCcnt), double buffered:
// stage kt+1 is in flight while WMMAs consume stage kt from LDS.
__global__ __launch_bounds__(256)
void gemm_xg_kernel(const bf16_t* __restrict__ X, const bf16_t* __restrict__ W,
                    const float* __restrict__ b0, const float* __restrict__ b1,
                    float* __restrict__ out) {
    __shared__ __align__(16) bf16_t a_lds[2][128 * 32];   // 2 x 8KB
    __shared__ __align__(16) bf16_t b_lds[2][64 * 32];    // 2 x 4KB

    const int tid  = threadIdx.x;
    const int lane = tid & 31;
    const int wave = tid >> 5;
    const int mBlock = blockIdx.y * 128;
    const int nBlock = blockIdx.x * 64;

    // staging pattern: 16B chunks, 4 per 32-element row
    const int ar0 = tid >> 2;                 // A rows 0..63
    const int ar1 = (tid + 256) >> 2;         // A rows 64..127
    const int aco = (tid & 3) * 8;            // chunk offset in elements
    const int br  = tid >> 2;                 // B rows 0..63

    const bf16_t* aG0 = X + (size_t)(mBlock + ar0) * EMB + aco;
    const bf16_t* aG1 = X + (size_t)(mBlock + ar1) * EMB + aco;
    const bf16_t* bG  = W + (size_t)(nBlock + br ) * EMB + aco;

    // fragment read addresses (LDS)
    const int ncol = lane & 15;
    const int k0a  = (lane >> 4) * 16;        // A fragment K base
    const int kb0  = (lane >> 4) * 16;        // B fragment K base
    const int arow = (wave * 16 + ncol) * 32;

    f32x8 acc[4];
    #pragma unroll
    for (int nt = 0; nt < 4; ++nt) acc[nt] = (f32x8){};

    // prologue: stage kt=0 into buffer 0 (3 async copies per thread)
    async_load_b128(&a_lds[0][ar0 * 32 + aco], aG0);
    async_load_b128(&a_lds[0][ar1 * 32 + aco], aG1);
    async_load_b128(&b_lds[0][br  * 32 + aco], bG);

    #pragma unroll
    for (int kt = 0; kt < EMB / 32; ++kt) {
        const int buf = kt & 1;
        if (kt < EMB / 32 - 1) {
            const int kb_n = (kt + 1) * 32;
            async_load_b128(&a_lds[buf ^ 1][ar0 * 32 + aco], aG0 + kb_n);
            async_load_b128(&a_lds[buf ^ 1][ar1 * 32 + aco], aG1 + kb_n);
            async_load_b128(&b_lds[buf ^ 1][br  * 32 + aco], bG  + kb_n);
            WAIT_ASYNC(3);       // stage kt landed; kt+1 still in flight
        } else {
            WAIT_ASYNC(0);
        }
        __syncthreads();

        union { bf16x16 v; bf16x8 h[2]; } a;
        a.h[0] = *(const bf16x8*)(&a_lds[buf][arow + k0a]);
        a.h[1] = *(const bf16x8*)(&a_lds[buf][arow + k0a + 8]);
        #pragma unroll
        for (int nt = 0; nt < 4; ++nt) {
            bf16x16 b = *(const bf16x16*)(&b_lds[buf][(nt * 16 + ncol) * 32 + kb0]);
            acc[nt] = wmma_bf16(a.v, b, acc[nt]);
        }
        __syncthreads();         // reads done before buf is overwritten
    }

    const int mOff = 8 * (lane >> 4);
    #pragma unroll
    for (int nt = 0; nt < 4; ++nt) {
        const int n = nBlock + nt * 16 + ncol;
        const float bias = b0[n] + b1[n];
        #pragma unroll
        for (int v = 0; v < 8; ++v) {
            const int m = mBlock + wave * 16 + mOff + v;
            out[(size_t)m * G4 + n] = acc[nt][v] + bias;
        }
    }
}

// ---------------------------------------------------------------- LSTM scan
// 2 persistent blocks (fwd / bwd), 1024 threads = 32 waves.
// Per step: g = xg[t] + h @ Whh^T via WMMA matvec (h broadcast in all A rows).
// xg[t+1] is prefetched into an LDS double buffer with async-to-LDS copies;
// W rows are register-prefetched at distance 2 to overlap L2 latency.
__global__ __launch_bounds__(1024)
void lstm_scan_kernel(const float* __restrict__ xg_f, const bf16_t* __restrict__ whh_f,
                      float* __restrict__ hf,
                      const float* __restrict__ xg_b, const bf16_t* __restrict__ whh_b,
                      float* __restrict__ hb) {
    const float*  xg = (blockIdx.x == 0) ? xg_f  : xg_b;
    const bf16_t* W  = (blockIdx.x == 0) ? whh_f : whh_b;
    float*        hs = (blockIdx.x == 0) ? hf    : hb;

    __shared__ __align__(16) bf16_t h_lds[HDIM];
    __shared__ float  g_lds[G4];
    __shared__ __align__(16) float  xg_lds[2][G4];

    const int tid  = threadIdx.x;
    const int lane = tid & 31;
    const int wave = tid >> 5;

    if (tid < HDIM) h_lds[tid] = (bf16_t)0.0f;
    float c_reg = 0.0f;

    // prologue: bring xg[0] into buffer 0 (512 lanes x 16B = 8KB)
    if (tid < 512) async_load_b128(&xg_lds[0][tid * 4], xg + tid * 4);
    WAIT_ASYNC(0);
    __syncthreads();

    const int ncol = lane & 15;
    const int k0a  = (lane >> 4) * 16;
    const int kb0  = (lane >> 4) * 16;

    for (int t = 0; t < SEQ; ++t) {
        // kick off prefetch of next step's gate pre-activations
        if (t + 1 < SEQ && tid < 512)
            async_load_b128(&xg_lds[(t + 1) & 1][tid * 4],
                            xg + (size_t)(t + 1) * G4 + tid * 4);

        const float* xgt = xg_lds[t & 1];
        #pragma unroll
        for (int tile = 0; tile < 4; ++tile) {
            const int nBase = (wave * 4 + tile) * 16;
            const bf16_t* wrow = W + (size_t)(nBase + ncol) * HDIM + kb0;
            f32x8 acc = (f32x8){};
            bf16x16 bq0 = *(const bf16x16*)(wrow);
            bf16x16 bq1 = *(const bf16x16*)(wrow + 32);
            #pragma unroll
            for (int j = 0; j < HDIM / 32; ++j) {
                bf16x16 bq2;
                if (j < HDIM / 32 - 2) bq2 = *(const bf16x16*)(wrow + 32 * (j + 2));
                union { bf16x16 v; bf16x8 h[2]; } a;
                a.h[0] = *(const bf16x8*)(h_lds + 32 * j + k0a);
                a.h[1] = *(const bf16x8*)(h_lds + 32 * j + k0a + 8);
                acc = wmma_bf16(a.v, bq0, acc);   // all D rows equal
                bq0 = bq1;
                if (j < HDIM / 32 - 2) bq1 = bq2;
            }
            if (lane < 16) g_lds[nBase + lane] = acc[0] + xgt[nBase + lane];
        }
        __syncthreads();
        if (tid < HDIM) {
            const float gi = g_lds[tid];
            const float gf = g_lds[HDIM + tid];
            const float gg = g_lds[2 * HDIM + tid];
            const float go = g_lds[3 * HDIM + tid];
            const float i = 1.0f / (1.0f + __expf(-gi));
            const float f = 1.0f / (1.0f + __expf(-gf));
            const float g = tanhf(gg);
            const float o = 1.0f / (1.0f + __expf(-go));
            c_reg = f * c_reg + i * g;
            const float h = o * tanhf(c_reg);
            hs[(size_t)t * HDIM + tid] = h;
            h_lds[tid] = (bf16_t)h;
        }
        WAIT_ASYNC(0);          // next step's xg tile landed in LDS
        __syncthreads();
    }
}

// ---------------------------------------------------------------- logits
// unary[c, t] = hf[t]·Wlin[c,:512] + hbrev[SEQ-1-t]·Wlin[c,512:] + blin[c]
__global__ __launch_bounds__(256)
void logits_kernel(const float* __restrict__ hf, const float* __restrict__ hbrev,
                   const float* __restrict__ Wlin, const float* __restrict__ blin,
                   float* __restrict__ unary, float* __restrict__ xbuf) {
    const int tid = threadIdx.x;
    const int t = blockIdx.x * 8 + (tid >> 5);
    const int c = tid & 31;
    const float* hfr = hf + (size_t)t * HDIM;
    const float* hbr = hbrev + (size_t)(SEQ - 1 - t) * HDIM;
    const float* w0  = Wlin + (size_t)c * (2 * HDIM);
    float s = blin[c];
    for (int j = 0; j < HDIM; ++j) s += hfr[j] * w0[j];
    for (int j = 0; j < HDIM; ++j) s += hbr[j] * w0[HDIM + j];
    unary[(size_t)c * SEQ + t] = s;
    xbuf[(size_t)c * SEQ + t]  = s;
}

// ---------------------------------------------------------------- CRF
__global__ void crf_softmax_kernel(const float* __restrict__ x, float* __restrict__ p) {
    const int l = blockIdx.x * 256 + threadIdx.x;
    float v[NCLS];
    float m = -1e30f;
    #pragma unroll
    for (int c = 0; c < NCLS; ++c) { v[c] = x[c * SEQ + l]; m = fmaxf(m, v[c]); }
    float s = 0.0f;
    #pragma unroll
    for (int c = 0; c < NCLS; ++c) { v[c] = __expf(v[c] - m); s += v[c]; }
    const float inv = 1.0f / s;
    #pragma unroll
    for (int c = 0; c < NCLS; ++c) p[c * SEQ + l] = v[c] * inv;
}

// x = unary + weight * (kern (*) p)   [comp = -I flips sign back to +]
__global__ void crf_update_kernel(const float* __restrict__ p, const float* __restrict__ unary,
                                  float* __restrict__ x,
                                  const float* __restrict__ wgt, const float* __restrict__ invt) {
    const int l = blockIdx.x * 256 + threadIdx.x;
    const int c = blockIdx.y;
    const float it = invt[0];
    const float w  = wgt[0];
    const float* pc = p + c * SEQ;
    float r = 0.0f;
    #pragma unroll
    for (int d = -5; d <= 5; ++d) {
        if (d == 0) continue;
        const float dd = (float)d * it;
        const float kv = __expf(-0.5f * dd * dd);
        const int ll = l + d;
        const float pv = (ll >= 0 && ll < SEQ) ? pc[ll] : 0.0f;
        r += kv * pv;
    }
    x[c * SEQ + l] = unary[c * SEQ + l] + w * r;
}

__global__ void crf_final_kernel(const float* __restrict__ x, float* __restrict__ out) {
    const int l = blockIdx.x * 256 + threadIdx.x;
    float v[NCLS];
    float m = -1e30f;
    #pragma unroll
    for (int c = 0; c < NCLS; ++c) { v[c] = x[c * SEQ + l]; m = fmaxf(m, v[c]); }
    float s = 0.0f;
    #pragma unroll
    for (int c = 0; c < NCLS; ++c) { v[c] = __expf(v[c] - m); s += v[c]; }
    const float inv = 1.0f / s;
    #pragma unroll
    for (int c = 0; c < NCLS; ++c) out[(size_t)l * NCLS + c] = v[c] * inv;
}

// ---------------------------------------------------------------- launch
extern "C" void kernel_launch(void* const* d_in, const int* in_sizes, int n_in,
                              void* d_out, int out_size, void* d_ws, size_t ws_size,
                              hipStream_t stream) {
    const float* batch  = (const float*)d_in[0];
    const float* W_ih_f = (const float*)d_in[1];
    const float* W_hh_f = (const float*)d_in[2];
    const float* b_ih_f = (const float*)d_in[3];
    const float* b_hh_f = (const float*)d_in[4];
    const float* W_ih_b = (const float*)d_in[5];
    const float* W_hh_b = (const float*)d_in[6];
    const float* b_ih_b = (const float*)d_in[7];
    const float* b_hh_b = (const float*)d_in[8];
    const float* W_lin  = (const float*)d_in[9];
    const float* b_lin  = (const float*)d_in[10];
    const float* wgt    = (const float*)d_in[11];
    const float* invt   = (const float*)d_in[12];

    char* ws = (char*)d_ws;
    size_t off = 0;
    auto carve = [&](size_t bytes) -> char* {
        char* p = ws + off;
        off += (bytes + 255) & ~(size_t)255;
        return p;
    };

    bf16_t* xb    = (bf16_t*)carve((size_t)SEQ * EMB * 2);
    bf16_t* xrev  = (bf16_t*)carve((size_t)SEQ * EMB * 2);
    bf16_t* wihf  = (bf16_t*)carve((size_t)G4 * EMB * 2);
    bf16_t* wihb  = (bf16_t*)carve((size_t)G4 * EMB * 2);
    bf16_t* whhf  = (bf16_t*)carve((size_t)G4 * HDIM * 2);
    bf16_t* whhb  = (bf16_t*)carve((size_t)G4 * HDIM * 2);
    float*  xgf   = (float*)carve((size_t)SEQ * G4 * 4);
    float*  xgb   = (float*)carve((size_t)SEQ * G4 * 4);
    float*  hf    = (float*)carve((size_t)SEQ * HDIM * 4);
    float*  hbrev = (float*)carve((size_t)SEQ * HDIM * 4);
    float*  unary = (float*)carve((size_t)NCLS * SEQ * 4);
    float*  xbuf  = (float*)carve((size_t)NCLS * SEQ * 4);
    float*  pbuf  = (float*)carve((size_t)NCLS * SEQ * 4);

    // 1) precision converts
    cvt_batch_kernel<<<(SEQ * EMB) / 256, 256, 0, stream>>>(batch, xb, xrev);
    const int nW = G4 * EMB;  // == G4 * HDIM
    cvt_f32_bf16_kernel<<<nW / 256, 256, 0, stream>>>(W_ih_f, wihf, nW);
    cvt_f32_bf16_kernel<<<nW / 256, 256, 0, stream>>>(W_ih_b, wihb, nW);
    cvt_f32_bf16_kernel<<<nW / 256, 256, 0, stream>>>(W_hh_f, whhf, nW);
    cvt_f32_bf16_kernel<<<nW / 256, 256, 0, stream>>>(W_hh_b, whhb, nW);

    // 2) input-projection GEMMs (WMMA, async-LDS pipelined)
    dim3 gdim(G4 / 64, SEQ / 128);
    gemm_xg_kernel<<<gdim, 256, 0, stream>>>(xb,   wihf, b_ih_f, b_hh_f, xgf);
    gemm_xg_kernel<<<gdim, 256, 0, stream>>>(xrev, wihb, b_ih_b, b_hh_b, xgb);

    // 3) sequential bidirectional scan (persistent, WMMA matvec + async LDS prefetch)
    lstm_scan_kernel<<<2, 1024, 0, stream>>>(xgf, whhf, hf, xgb, whhb, hbrev);

    // 4) output projection
    logits_kernel<<<SEQ / 8, 256, 0, stream>>>(hf, hbrev, W_lin, b_lin, unary, xbuf);

    // 5) mean-field CRF iterations
    for (int it = 0; it < NIT; ++it) {
        crf_softmax_kernel<<<SEQ / 256, 256, 0, stream>>>(xbuf, pbuf);
        crf_update_kernel<<<dim3(SEQ / 256, NCLS), 256, 0, stream>>>(pbuf, unary, xbuf, wgt, invt);
    }
    crf_final_kernel<<<SEQ / 256, 256, 0, stream>>>(xbuf, (float*)d_out);
}
